// EquiSite_48137993454081
// MI455X (gfx1250) — matrix operand order repaired and checked by reading, not compiled, not run on hardware
//
#include <hip/hip_runtime.h>
#include <math.h>

typedef __attribute__((ext_vector_type(16))) _Float16 v16h;
typedef __attribute__((ext_vector_type(8)))  float    v8f;

#define HID 128
#define NRAD 6
#define CUTF 11.5f
#define PI_F 3.14159265358979323846f

struct F3 { float x, y, z; };
__device__ __forceinline__ F3 f3sub(F3 a, F3 b){ return {a.x-b.x, a.y-b.y, a.z-b.z}; }
__device__ __forceinline__ F3 f3cross(F3 a, F3 b){
  return {a.y*b.z-a.z*b.y, a.z*b.x-a.x*b.z, a.x*b.y-a.y*b.x};
}
__device__ __forceinline__ float f3dot(F3 a, F3 b){ return a.x*b.x + a.y*b.y + a.z*b.z; }
__device__ __forceinline__ float f3norm(F3 a){ return sqrtf(f3dot(a,a)); }
__device__ __forceinline__ F3 ldpos(const float* p, int idx){
  return {p[3*idx+0], p[3*idx+1], p[3*idx+2]};
}
__device__ __forceinline__ float swishf(float v){ return v / (1.f + __expf(-v)); }
__device__ __forceinline__ v8f v8f_zero(){
  v8f v;
#pragma unroll
  for (int i = 0; i < 8; ++i) v[i] = 0.f;
  return v;
}

// ---------------------------------------------------------------------------
// Weight transpose + f32->f16 convert:  Wt[n*Kpad + k] = W[k*Nout + n]
// ---------------------------------------------------------------------------
__global__ __launch_bounds__(256) void convw_kernel(
    const float* __restrict__ W, _Float16* __restrict__ Wt,
    int Kin, int Kpad, int Nout)
{
  int idx = blockIdx.x * blockDim.x + threadIdx.x;
  if (idx >= Nout * Kpad) return;
  int n = idx / Kpad;
  int k = idx - n * Kpad;
  Wt[idx] = (k < Kin) ? (_Float16)W[k * Nout + n] : (_Float16)0.f;
}

// ---------------------------------------------------------------------------
// x[:, 0:64] = [one_hot(z) | bb | sc] @ W_emb + b_emb     (40x64, scalar path)
// ---------------------------------------------------------------------------
__global__ __launch_bounds__(256) void node_emb_kernel(
    const int* __restrict__ z, const float* __restrict__ bb,
    const float* __restrict__ scb, const float* __restrict__ W_emb,
    const float* __restrict__ b_emb, float* __restrict__ x, int Nn)
{
  int idx = blockIdx.x * blockDim.x + threadIdx.x;
  if (idx >= Nn * 64) return;
  int n = idx >> 6, c = idx & 63;
  float acc = b_emb[c] + W_emb[z[n] * 64 + c];
#pragma unroll
  for (int k = 0; k < 6; ++k) acc += bb[n*6 + k] * W_emb[(26 + k)*64 + c];
#pragma unroll
  for (int k = 0; k < 8; ++k) acc += scb[n*8 + k] * W_emb[(32 + k)*64 + c];
  x[(size_t)n * HID + c] = acc;
}

// ---------------------------------------------------------------------------
// x[:, 64:128] = esm @ W_esm + b_esm   — WMMA, K=1280, one wave per 16 rows.
// A-fragment converted f32->f16 straight from global (streaming 256 MB read).
// ---------------------------------------------------------------------------
__global__ __launch_bounds__(128) void node_esm_kernel(
    const float* __restrict__ esm, const _Float16* __restrict__ Wesmt,
    const float* __restrict__ b_esm, float* __restrict__ x, int Nn)
{
  const int lane = threadIdx.x & 31;
  const int wv   = threadIdx.x >> 5;
  const int rowTiles = (Nn + 15) >> 4;
  const int tile = blockIdx.x * 4 + wv;
  if (tile >= rowTiles) return;
  const int rbase = tile << 4;
  const int ln = lane & 15;
  const int hi = lane >> 4;
  int arow = rbase + ln; if (arow >= Nn) arow = Nn - 1;
  const float* ap = esm + (size_t)arow * 1280;

  v8f acc[4];
#pragma unroll
  for (int c = 0; c < 4; ++c) acc[c] = v8f_zero();

  for (int kc = 0; kc < 40; ++kc) {
    const int kb = kc * 32;
    __builtin_prefetch(ap + kb + 128, 0, 0);
    v16h a;
#pragma unroll
    for (int t = 0; t < 8; ++t) {
      a[t]     = (_Float16)ap[kb + hi*16 + t];
      a[t + 8] = (_Float16)ap[kb + hi*16 + 8 + t];
    }
#pragma unroll
    for (int c = 0; c < 4; ++c) {
      v16h b = *(const v16h*)(Wesmt + (size_t)(c*16 + ln) * 1280 + kb + hi*16);
      acc[c] = __builtin_amdgcn_wmma_f32_16x16x32_f16(
          false, a, false, b, (short)0, acc[c], false, false);
    }
  }

  if (rbase + 16 <= Nn) {            // wave-uniform fast path: full tile
#pragma unroll
    for (int c = 0; c < 4; ++c) {
      const int col = c*16 + ln;
      const float bias = b_esm[col];
      float* xp = x + (size_t)(rbase + hi*8) * HID + 64 + col;
#pragma unroll
      for (int r = 0; r < 8; ++r) xp[(size_t)r * HID] = acc[c][r] + bias;
    }
  } else {
#pragma unroll
    for (int c = 0; c < 4; ++c) {
      const int col = c*16 + ln;
      const float bias = b_esm[col];
#pragma unroll
      for (int r = 0; r < 8; ++r) {
        const int row = rbase + hi*8 + r;
        if (row < Nn) x[(size_t)row * HID + 64 + col] = acc[c][r] + bias;
      }
    }
  }
}

// ---------------------------------------------------------------------------
// Edge kernel: one wave per 16-edge tile.
//   lanes 0..15: full geometry -> 124 f16 features into LDS
//   all lanes : WMMA feat@W_edge and x[j]@W_msg, m = swish*swish, atomic agg
// ---------------------------------------------------------------------------
__global__ __launch_bounds__(128) void edge_kernel(
    const float* __restrict__ pos, const float* __restrict__ pos_n,
    const float* __restrict__ pos_c, const int* __restrict__ eidx,
    const float* __restrict__ x, const _Float16* __restrict__ Wedget,
    const _Float16* __restrict__ Wmsgt, float* __restrict__ agg,
    int Nn, int Ee)
{
  __shared__ _Float16 featS[4][16][136];   // padded row stride vs bank conflicts
  __shared__ int jS[4][16];
  __shared__ int iS[4][16];

  const int lane = threadIdx.x & 31;
  const int wv   = threadIdx.x >> 5;
  const int nTiles = (Ee + 15) >> 4;
  const int tile = blockIdx.x * 4 + wv;
  const bool active = tile < nTiles;
  const int ebase = tile << 4;

  if (active && lane < 16) {
    const int e = ebase + lane;
    _Float16* f = &featS[wv][lane][0];
    if (e < Ee) {
      const int j = eidx[e];
      const int i = eidx[Ee + e];
      jS[wv][lane] = j; iS[wv][lane] = i;
      F3 Pi = ldpos(pos, i), Pj = ldpos(pos, j);
      F3 vji = f3sub(Pj, Pi);
      float dist = f3norm(vji);
      int r0 = i - 1; if (r0 < 0) r0 += Nn;
      int r1 = i + 1; if (r1 >= Nn) r1 -= Nn;
      F3 vr0 = f3sub(ldpos(pos, r0), Pi);
      F3 vr1 = f3sub(ldpos(pos, r1), Pi);
      float theta = atan2f(f3norm(f3cross(vji, vr0)), f3dot(vji, vr0));
      F3 p1 = f3cross(vr0, vr1);
      F3 p2 = f3cross(vr0, vji);
      float pb = f3dot(f3cross(p1, p2), vr0) / (f3norm(vr0) + 1e-7f);
      float phi = atan2f(pb, f3dot(p1, p2));
      // backbone frame Euler angles
      F3 o1x = f3sub(ldpos(pos_n, i), Pi);
      F3 o1z = f3cross(o1x, f3cross(o1x, f3sub(ldpos(pos_c, i), Pi)));
      float o1zl = f3norm(o1z) + 1e-7f;
      F3 o2x = f3sub(ldpos(pos_n, j), Pj);
      F3 o2z = f3cross(o2x, f3cross(o2x, f3sub(ldpos(pos_c, j), Pj)));
      float o2zl = f3norm(o2z) + 1e-7f;
      F3 nv = f3cross(o1z, o2z);
      float a1 = atan2f(f3dot(f3cross(o1x, nv), o1z) / o1zl, f3dot(o1x, nv));
      float a2 = atan2f(f3norm(nv), f3dot(o1z, o2z));
      float a3 = atan2f(f3dot(f3cross(nv, o2x), o2z) / o2zl, f3dot(nv, o2x));
      // radial basis with cosine cutoff
      float u = fminf(fmaxf(dist / CUTF, 0.f), 1.f);
      float fc = 0.5f * (cosf(PI_F * u) + 1.f);
      float sc = sqrtf(2.f / CUTF) * fc / (dist + 1e-6f);
      float rb[NRAD];
#pragma unroll
      for (int n = 0; n < NRAD; ++n) rb[n] = sc * sinf((float)(n + 1) * PI_F * u);
      // feature0: [NR,3,3] -> idx = n*9 + l*3 + m  (l: theta, m: phi)
      float ct[3] = {1.f, cosf(theta), cosf(2.f * theta)};
      float cp[3] = {1.f, cosf(phi),   cosf(2.f * phi)};
#pragma unroll
      for (int n = 0; n < NRAD; ++n)
#pragma unroll
        for (int l = 0; l < 3; ++l)
#pragma unroll
          for (int m = 0; m < 3; ++m)
            f[n*9 + l*3 + m] = (_Float16)(rb[n] * ct[l] * cp[m]);
      // feature1: three angle embeddings, each [NR,3] -> idx = n*3 + l
      float ang3[3] = {a1, a2, a3};
#pragma unroll
      for (int q = 0; q < 3; ++q) {
        float ca[3] = {1.f, cosf(ang3[q]), cosf(2.f * ang3[q])};
#pragma unroll
        for (int n = 0; n < NRAD; ++n)
#pragma unroll
          for (int l = 0; l < 3; ++l)
            f[54 + q*18 + n*3 + l] = (_Float16)(rb[n] * ca[l]);
      }
      // sinusoidal positional embedding of sequence offset (NPE=16)
      float dpe = (float)(j - i);
#pragma unroll
      for (int k = 0; k < 8; ++k) {
        float fr = __expf(-(float)(2*k) * 0.57564627324851148f); // ln(1e4)/16
        float an = dpe * fr;
        f[108 + k] = (_Float16)cosf(an);
        f[116 + k] = (_Float16)sinf(an);
      }
      f[124] = (_Float16)0.f; f[125] = (_Float16)0.f;
      f[126] = (_Float16)0.f; f[127] = (_Float16)0.f;
    } else {
      jS[wv][lane] = 0; iS[wv][lane] = 0;
      for (int k = 0; k < 128; ++k) f[k] = (_Float16)0.f;
    }
  }
  __syncthreads();
  if (!active) return;

  const int ln = lane & 15;
  const int hi = lane >> 4;
  // A fragments: feat tile from LDS, x[j] tile straight from (L2-hot) global
  v16h fa[4], xa[4];
  {
    const _Float16* fp = &featS[wv][ln][0];
    const float* xp = x + (size_t)jS[wv][ln] * HID;
#pragma unroll
    for (int kc = 0; kc < 4; ++kc) {
      const int kb = kc * 32 + hi * 16;
#pragma unroll
      for (int t = 0; t < 8; ++t) {
        fa[kc][t]     = fp[kb + t];
        fa[kc][t + 8] = fp[kb + 8 + t];
        xa[kc][t]     = (_Float16)xp[kb + t];
        xa[kc][t + 8] = (_Float16)xp[kb + 8 + t];
      }
    }
  }
  const bool fullTile = (ebase + 16 <= Ee);   // wave-uniform
#pragma unroll
  for (int c = 0; c < 8; ++c) {
    v8f accE = v8f_zero();
    v8f accM = v8f_zero();
    const int nOut = c * 16 + ln;
#pragma unroll
    for (int kc = 0; kc < 4; ++kc) {
      const int kb = kc * 32 + hi * 16;
      v16h bE = *(const v16h*)(Wedget + (size_t)nOut * HID + kb);
      v16h bM = *(const v16h*)(Wmsgt  + (size_t)nOut * HID + kb);
      accE = __builtin_amdgcn_wmma_f32_16x16x32_f16(
          false, fa[kc], false, bE, (short)0, accE, false, false);
      accM = __builtin_amdgcn_wmma_f32_16x16x32_f16(
          false, xa[kc], false, bM, (short)0, accM, false, false);
    }
    if (fullTile) {                  // fast path: no per-element EXEC churn
#pragma unroll
      for (int r = 0; r < 8; ++r) {
        float m = swishf(accM[r]) * swishf(accE[r]);
        atomicAdd(&agg[(size_t)iS[wv][hi*8 + r] * HID + nOut], m);
      }
    } else {
#pragma unroll
      for (int r = 0; r < 8; ++r) {
        const int row = hi * 8 + r;
        if (ebase + row < Ee) {
          float m = swishf(accM[r]) * swishf(accE[r]);
          atomicAdd(&agg[(size_t)iS[wv][row] * HID + nOut], m);
        }
      }
    }
  }
}

// ---------------------------------------------------------------------------
// xe = x + swish(agg @ W_upd)   — WMMA, K=128, one wave per 16 rows
// ---------------------------------------------------------------------------
__global__ __launch_bounds__(128) void node_update_kernel(
    const float* __restrict__ agg, const _Float16* __restrict__ Wupdt,
    const float* __restrict__ x, float* __restrict__ xe, int Nn)
{
  const int lane = threadIdx.x & 31;
  const int wv   = threadIdx.x >> 5;
  const int rowTiles = (Nn + 15) >> 4;
  const int tile = blockIdx.x * 4 + wv;
  if (tile >= rowTiles) return;
  const int rbase = tile << 4;
  const int ln = lane & 15;
  const int hi = lane >> 4;
  int arow = rbase + ln; if (arow >= Nn) arow = Nn - 1;
  const float* ap = agg + (size_t)arow * HID;

  v16h a[4];
#pragma unroll
  for (int kc = 0; kc < 4; ++kc) {
    const int kb = kc * 32 + hi * 16;
#pragma unroll
    for (int t = 0; t < 8; ++t) {
      a[kc][t]     = (_Float16)ap[kb + t];
      a[kc][t + 8] = (_Float16)ap[kb + 8 + t];
    }
  }
  const bool fullTile = (rbase + 16 <= Nn);   // wave-uniform
#pragma unroll
  for (int c = 0; c < 8; ++c) {
    v8f acc = v8f_zero();
#pragma unroll
    for (int kc = 0; kc < 4; ++kc) {
      v16h b = *(const v16h*)(Wupdt + (size_t)(c*16 + ln) * HID + kc*32 + hi*16);
      acc = __builtin_amdgcn_wmma_f32_16x16x32_f16(
          false, a[kc], false, b, (short)0, acc, false, false);
    }
    const int col = c*16 + ln;
    if (fullTile) {
      const float* xp  = x  + (size_t)(rbase + hi*8) * HID + col;
      float*       xep = xe + (size_t)(rbase + hi*8) * HID + col;
#pragma unroll
      for (int r = 0; r < 8; ++r)
        xep[(size_t)r * HID] = xp[(size_t)r * HID] + swishf(acc[r]);
    } else {
#pragma unroll
      for (int r = 0; r < 8; ++r) {
        const int row = rbase + hi*8 + r;
        if (row < Nn)
          xe[(size_t)row * HID + col] = x[(size_t)row * HID + col] + swishf(acc[r]);
      }
    }
  }
}

// ---------------------------------------------------------------------------
// Output heads: one wave per node (small GEMVs, scalar VALU; ~1 GFLOP total)
//   out[node]            = sigmoid(relu(relu(xe@Wo0+b)@Wo1+b)@Wof+b)
//   out[Nn + 2*node + d] = relu(x@Wn0+b) @ Wnf + b
// ---------------------------------------------------------------------------
__global__ __launch_bounds__(256) void heads_kernel(
    const float* __restrict__ x,  const float* __restrict__ xe,
    const float* __restrict__ W_o0, const float* __restrict__ b_o0,
    const float* __restrict__ W_o1, const float* __restrict__ b_o1,
    const float* __restrict__ W_of, const float* __restrict__ b_of,
    const float* __restrict__ W_n0, const float* __restrict__ b_n0,
    const float* __restrict__ W_nf, const float* __restrict__ b_nf,
    float* __restrict__ out, int Nn)
{
  __shared__ float xS[8][128], xeS[8][128], h0S[8][128], h1S[8][32], xnS[8][32];
  const int lane = threadIdx.x & 31;
  const int wv   = threadIdx.x >> 5;
  int node = blockIdx.x * 8 + wv;
  const bool real = node < Nn;
  if (!real) node = Nn - 1;

#pragma unroll
  for (int t = 0; t < 4; ++t) {
    int col = lane + t * 32;
    xS [wv][col] = x [(size_t)node * HID + col];
    xeS[wv][col] = xe[(size_t)node * HID + col];
  }
  __syncthreads();

#pragma unroll
  for (int t = 0; t < 4; ++t) {
    int c = lane + t * 32;
    float h = b_o0[c];
    for (int k = 0; k < 128; ++k) h += xeS[wv][k] * W_o0[k * 128 + c];
    h0S[wv][c] = fmaxf(h, 0.f);
  }
  {
    float h = b_n0[lane];
    for (int k = 0; k < 128; ++k) h += xS[wv][k] * W_n0[k * 32 + lane];
    xnS[wv][lane] = fmaxf(h, 0.f);
  }
  __syncthreads();
  {
    float h = b_o1[lane];
    for (int k = 0; k < 128; ++k) h += h0S[wv][k] * W_o1[k * 32 + lane];
    h1S[wv][lane] = fmaxf(h, 0.f);
  }
  __syncthreads();

  if (lane == 0 && real) {
    float o = b_of[0];
    for (int k = 0; k < 32; ++k) o += h1S[wv][k] * W_of[k];
    out[node] = 1.f / (1.f + __expf(-o));
    float n0 = b_nf[0], n1 = b_nf[1];
    for (int k = 0; k < 32; ++k) {
      n0 += xnS[wv][k] * W_nf[k*2 + 0];
      n1 += xnS[wv][k] * W_nf[k*2 + 1];
    }
    out[Nn + (size_t)node * 2 + 0] = n0;
    out[Nn + (size_t)node * 2 + 1] = n1;
  }
}

// ---------------------------------------------------------------------------
extern "C" void kernel_launch(void* const* d_in, const int* in_sizes, int n_in,
                              void* d_out, int out_size, void* d_ws, size_t ws_size,
                              hipStream_t stream)
{
  const int*   z      = (const int*)  d_in[0];
  const float* pos    = (const float*)d_in[1];
  const float* pos_n  = (const float*)d_in[2];
  const float* pos_c  = (const float*)d_in[3];
  const float* bb     = (const float*)d_in[4];
  const float* scb    = (const float*)d_in[5];
  const float* esm    = (const float*)d_in[6];
  /* d_in[7] = batch (unused, all zeros) */
  const int*   eidx   = (const int*)  d_in[8];
  const float* W_emb  = (const float*)d_in[9];
  const float* b_emb  = (const float*)d_in[10];
  const float* W_esm  = (const float*)d_in[11];
  const float* b_esm  = (const float*)d_in[12];
  const float* W_edge = (const float*)d_in[13];
  const float* W_msg  = (const float*)d_in[14];
  const float* W_upd  = (const float*)d_in[15];
  const float* W_o0   = (const float*)d_in[16];
  const float* b_o0   = (const float*)d_in[17];
  const float* W_o1   = (const float*)d_in[18];
  const float* b_o1   = (const float*)d_in[19];
  const float* W_of   = (const float*)d_in[20];
  const float* b_of   = (const float*)d_in[21];
  const float* W_n0   = (const float*)d_in[22];
  const float* b_n0   = (const float*)d_in[23];
  const float* W_nf   = (const float*)d_in[24];
  const float* b_nf   = (const float*)d_in[25];
  (void)n_in; (void)out_size; (void)ws_size;

  const int Nn = in_sizes[0];          // 50000
  const int Ee = in_sizes[8] / 2;      // 1000000

  // -------- workspace carve (256B aligned) --------
  char* ws = (char*)d_ws;
  size_t off = 0;
  auto carve = [&](size_t bytes) -> void* {
    void* p = ws + off;
    off += (bytes + 255) & ~(size_t)255;
    return p;
  };
  _Float16* Wesmt  = (_Float16*)carve((size_t)64  * 1280 * sizeof(_Float16));
  _Float16* Wedget = (_Float16*)carve((size_t)128 * 128  * sizeof(_Float16));
  _Float16* Wmsgt  = (_Float16*)carve((size_t)128 * 128  * sizeof(_Float16));
  _Float16* Wupdt  = (_Float16*)carve((size_t)128 * 128  * sizeof(_Float16));
  float* xbuf = (float*)carve((size_t)Nn * HID * sizeof(float));
  float* agg  = (float*)carve((size_t)Nn * HID * sizeof(float));
  float* xe   = (float*)carve((size_t)Nn * HID * sizeof(float));

  hipMemsetAsync(agg, 0, (size_t)Nn * HID * sizeof(float), stream);

  // weights -> transposed f16 [N][Kpad]
  convw_kernel<<<(64*1280 + 255)/256, 256, 0, stream>>>(W_esm,  Wesmt, 1280, 1280, 64);
  convw_kernel<<<(128*128 + 255)/256, 256, 0, stream>>>(W_edge, Wedget, 124, 128, 128);
  convw_kernel<<<(128*128 + 255)/256, 256, 0, stream>>>(W_msg,  Wmsgt,  128, 128, 128);
  convw_kernel<<<(128*128 + 255)/256, 256, 0, stream>>>(W_upd,  Wupdt,  128, 128, 128);

  // node embedding
  node_emb_kernel<<<((size_t)Nn*64 + 255)/256, 256, 0, stream>>>(
      z, bb, scb, W_emb, b_emb, xbuf, Nn);
  const int rowTiles = (Nn + 15) / 16;
  node_esm_kernel<<<(rowTiles + 3)/4, 128, 0, stream>>>(esm, Wesmt, b_esm, xbuf, Nn);

  // edge message passing (geometry + WMMA + atomic scatter)
  const int eTiles = (Ee + 15) / 16;
  edge_kernel<<<(eTiles + 3)/4, 128, 0, stream>>>(
      pos, pos_n, pos_c, eidx, xbuf, Wedget, Wmsgt, agg, Nn, Ee);

  // node update
  node_update_kernel<<<(rowTiles + 3)/4, 128, 0, stream>>>(agg, Wupdt, xbuf, xe, Nn);

  // heads -> d_out = [out (Nn), node_out (Nn*2)]
  heads_kernel<<<(Nn + 7)/8, 256, 0, stream>>>(
      xbuf, xe, W_o0, b_o0, W_o1, b_o1, W_of, b_of, W_n0, b_n0, W_nf, b_nf,
      (float*)d_out, Nn);
}
